// TiledAttention_65687229825650
// MI455X (gfx1250) — matrix operand order repaired and checked by reading, not compiled, not run on hardware
//
#include <hip/hip_runtime.h>
#include <hip/hip_bf16.h>

typedef __attribute__((ext_vector_type(16))) _Float16 v16h;
typedef __attribute__((ext_vector_type(8)))  _Float16 h8;
typedef __attribute__((ext_vector_type(8)))  float    v8f;

#define WMMA_F32_F16(A, B, C) \
  __builtin_amdgcn_wmma_f32_16x16x32_f16(false, (A), false, (B), (short)0, (C), false, false)

__device__ __forceinline__ v16h cat8(h8 lo, h8 hi) {
  return __builtin_shufflevector(lo, hi, 0,1,2,3,4,5,6,7,8,9,10,11,12,13,14,15);
}

static constexpr int T  = 2048;
static constexpr int D  = 1024;
static constexpr int H  = 16;
static constexpr int Dh = 64;

// ---------------------------------------------------------------------------
// Kernel 1: fused Q/K/V projection.  C = x @ W^T (NT GEMM, k contiguous on
// both sides).  128x128 block tile, 8 waves of 32x64, K-step 64 (16 WMMAs per
// stage), register-staged double-buffered LDS so global-load waits land after
// the WMMA burst.  Epilogues:
//   z==0: Q * (1/sqrt(Dh)) f16 [b,h,t,dh]      (scale folded here)
//   z==1: K f16              [b,h,t,dh]
//   z==2: V transposed f16   [b,h,dh,t]  -> packed b128 stores (t contiguous)
// ---------------------------------------------------------------------------
__global__ __launch_bounds__(256)
void proj_qkv_kernel(const float* __restrict__ x,
                     const float* __restrict__ Wq,
                     const float* __restrict__ Wk,
                     const float* __restrict__ Wv,
                     _Float16* __restrict__ Qo,
                     _Float16* __restrict__ Ko,
                     _Float16* __restrict__ Vto) {
  __shared__ __align__(16) _Float16 As[2][128 * 64];   // 2 x 16 KB
  __shared__ __align__(16) _Float16 Ws[2][128 * 64];   // 2 x 16 KB

  const int n0 = blockIdx.x * 128;
  const int m0 = blockIdx.y * 128;
  const int z  = blockIdx.z;
  const float* W = (z == 0) ? Wq : (z == 1) ? Wk : Wv;

  const int tid  = threadIdx.x;
  const int lane = tid & 31;
  const int wid  = tid >> 5;
  const int rm   = (wid >> 1) * 32;
  const int cn   = (wid & 1) * 64;
  const int hi16 = (lane >= 16) ? 1 : 0;
  const int ln   = lane & 15;

  v8f acc[2][4];
  #pragma unroll
  for (int i = 0; i < 2; ++i)
    #pragma unroll
    for (int j = 0; j < 4; ++j)
      acc[i][j] = v8f{0.f,0.f,0.f,0.f,0.f,0.f,0.f,0.f};

  // ---- register staging for one 128x64 A tile + one 128x64 W tile ----
  float4 ra[8], rw[8];

  #define FETCH_TILES(K0)                                                     \
    {                                                                         \
      _Pragma("unroll")                                                       \
      for (int it = 0; it < 8; ++it) {                                        \
        const int idx = tid + it * 256;            /* 2048 float4 units */    \
        const int row = idx >> 4;                                             \
        const int c4  = (idx & 15) * 4;                                       \
        ra[it] = *(const float4*)(x + (size_t)(m0 + row) * D + (K0) + c4);    \
        rw[it] = *(const float4*)(W + (size_t)(n0 + row) * D + (K0) + c4);    \
      }                                                                       \
    }

  #define STORE_TILES(P)                                                      \
    {                                                                         \
      _Pragma("unroll")                                                       \
      for (int it = 0; it < 8; ++it) {                                        \
        const int idx = tid + it * 256;                                       \
        const int row = idx >> 4;                                             \
        const int c4  = (idx & 15) * 4;                                       \
        _Float16* da = &As[P][row * 64 + c4];                                 \
        da[0] = (_Float16)ra[it].x; da[1] = (_Float16)ra[it].y;               \
        da[2] = (_Float16)ra[it].z; da[3] = (_Float16)ra[it].w;               \
        _Float16* dw = &Ws[P][row * 64 + c4];                                 \
        dw[0] = (_Float16)rw[it].x; dw[1] = (_Float16)rw[it].y;               \
        dw[2] = (_Float16)rw[it].z; dw[3] = (_Float16)rw[it].w;               \
      }                                                                       \
    }

  #define COMPUTE_STAGE(P)                                                    \
    {                                                                         \
      v16h afrag[2][2];                                                       \
      _Pragma("unroll")                                                       \
      for (int mt = 0; mt < 2; ++mt) {                                        \
        const _Float16* rp = &As[P][(rm + mt * 16 + ln) * 64];                \
        _Pragma("unroll")                                                     \
        for (int kc = 0; kc < 2; ++kc) {                                      \
          const int base = kc * 32 + (hi16 ? 8 : 0);                          \
          afrag[mt][kc] = cat8(*(const h8*)(rp + base),                       \
                               *(const h8*)(rp + base + 16));                 \
        }                                                                     \
      }                                                                       \
      _Pragma("unroll")                                                       \
      for (int kc = 0; kc < 2; ++kc)                                          \
        _Pragma("unroll")                                                     \
        for (int nt = 0; nt < 4; ++nt) {                                      \
          const _Float16* rp = &Ws[P][(cn + nt * 16 + ln) * 64];              \
          const int base = kc * 32 + (hi16 ? 8 : 0);                          \
          v16h bfrag = cat8(*(const h8*)(rp + base),                          \
                            *(const h8*)(rp + base + 16));                    \
          acc[0][nt] = WMMA_F32_F16(afrag[0][kc], bfrag, acc[0][nt]);         \
          acc[1][nt] = WMMA_F32_F16(afrag[1][kc], bfrag, acc[1][nt]);         \
        }                                                                     \
    }

  FETCH_TILES(0);
  STORE_TILES(0);
  __syncthreads();

  int p = 0;
  for (int k0 = 0;;) {
    const bool more = (k0 + 64) < D;
    if (more) FETCH_TILES(k0 + 64);   // global loads issued before WMMA burst
    COMPUTE_STAGE(p);                 // waits only on LDS of current buffer
    if (!more) break;
    STORE_TILES(p ^ 1);               // convert+spill next tile behind compute
    __syncthreads();
    p ^= 1;
    k0 += 64;
  }

  // ---- epilogue: uniform z branch hoisted out of the store loops ----
  if (z == 2) {
    // Vt[b,h,dh,t]: the 8 accum rows are consecutive t -> one b128 per tile.
    #pragma unroll
    for (int mt = 0; mt < 2; ++mt)
      #pragma unroll
      for (int nt = 0; nt < 4; ++nt) {
        h8 hv;
        #pragma unroll
        for (int r = 0; r < 8; ++r) hv[r] = (_Float16)acc[mt][nt][r];
        const int mb = m0 + rm + mt * 16 + hi16 * 8;   // first t of the run
        const int n  = n0 + cn + nt * 16 + ln;
        const int b = mb >> 11, t = mb & (T - 1);
        const int hh = n >> 6, dh = n & 63;
        *(h8*)&Vto[((size_t)((b * H + hh) * Dh + dh)) * T + t] = hv;
      }
  } else {
    _Float16* dst = (z == 0) ? Qo : Ko;
    const float sc = (z == 0) ? 0.125f : 1.0f;
    #pragma unroll
    for (int mt = 0; mt < 2; ++mt)
      #pragma unroll
      for (int nt = 0; nt < 4; ++nt)
        #pragma unroll
        for (int r = 0; r < 8; ++r) {
          const int m = m0 + rm + mt * 16 + r + hi16 * 8;
          const int n = n0 + cn + nt * 16 + ln;
          const int b = m >> 11, t = m & (T - 1);
          const int hh = n >> 6, dh = n & 63;
          dst[((size_t)((b * H + hh) * T + t)) * Dh + dh] =
              (_Float16)(acc[mt][nt][r] * sc);
        }
  }
  #undef FETCH_TILES
  #undef STORE_TILES
  #undef COMPUTE_STAGE
}

// ---------------------------------------------------------------------------
// Kernel 2: causal flash attention.  1 block = 64 queries of one (b,h);
// 4 waves, each wave owns 16 query rows entirely in registers.
// Per 32-wide kv tile: 4 WMMAs for Q.K^T, online softmax with half-wave
// shuffle reductions, P via per-wave LDS scratch -> A-fragment, 4 WMMAs P.V.
// Softmax scale pre-folded into Q; next kv tile prefetched.
// ---------------------------------------------------------------------------
__global__ __launch_bounds__(128)
void flash_attn_kernel(const _Float16* __restrict__ Q,
                       const _Float16* __restrict__ K,
                       const _Float16* __restrict__ Vt,
                       _Float16* __restrict__ O16) {
  __shared__ __align__(16) _Float16 Ps[4][16 * 32];

  const int bh   = blockIdx.x >> 5;    // b*H + h  (0..31)
  const int qblk = blockIdx.x & 31;
  const int tid  = threadIdx.x;
  const int lane = tid & 31;
  const int wid  = tid >> 5;
  const int hi16 = (lane >= 16) ? 1 : 0;
  const int ln   = lane & 15;

  const int q0 = qblk * 64 + wid * 16;
  const _Float16* Qb = Q  + (size_t)bh * T * Dh;
  const _Float16* Kb = K  + (size_t)bh * T * Dh;
  const _Float16* Vb = Vt + (size_t)bh * Dh * T;

  v16h qfrag[2];
  {
    const _Float16* rp = Qb + (size_t)(q0 + ln) * Dh;
    #pragma unroll
    for (int c = 0; c < 2; ++c) {
      const int base = c * 32 + (hi16 ? 8 : 0);
      qfrag[c] = cat8(*(const h8*)(rp + base), *(const h8*)(rp + base + 16));
    }
  }

  float Mr[8], Lr[8];
  v8f Oacc[4];
  #pragma unroll
  for (int r = 0; r < 8; ++r) { Mr[r] = -1e30f; Lr[r] = 0.f; }
  #pragma unroll
  for (int d = 0; d < 4; ++d) Oacc[d] = v8f{0.f,0.f,0.f,0.f,0.f,0.f,0.f,0.f};

  _Float16* myP = &Ps[wid][0];
  const int kvmax = q0 + 15;

  for (int kv0 = 0; kv0 <= kvmax; kv0 += 32) {
    // prefetch next kv tile's K and V streams (global_prefetch_b8)
    if (kv0 + 32 <= kvmax) {
      __builtin_prefetch(Kb + (size_t)(kv0 + 32 + lane) * Dh, 0, 3);
      __builtin_prefetch(Vb + (size_t)lane * T + kv0 + 32, 0, 3);
    }

    // ---- S = Q . K^T ----
    v8f s[2];
    #pragma unroll
    for (int nt = 0; nt < 2; ++nt) {
      s[nt] = v8f{0.f,0.f,0.f,0.f,0.f,0.f,0.f,0.f};
      const _Float16* kp = Kb + (size_t)(kv0 + nt * 16 + ln) * Dh;
      #pragma unroll
      for (int c = 0; c < 2; ++c) {
        const int base = c * 32 + (hi16 ? 8 : 0);
        v16h bfrag = cat8(*(const h8*)(kp + base), *(const h8*)(kp + base + 16));
        s[nt] = WMMA_F32_F16(qfrag[c], bfrag, s[nt]);
      }
    }

    // ---- causal mask ----
    #pragma unroll
    for (int nt = 0; nt < 2; ++nt)
      #pragma unroll
      for (int r = 0; r < 8; ++r) {
        const int qi = q0 + r + hi16 * 8;
        const int ki = kv0 + nt * 16 + ln;
        if (ki > qi) s[nt][r] = -1e30f;
      }

    // ---- online softmax: half-wave reductions over the 16 n lanes ----
    float alpha[8];
    #pragma unroll
    for (int r = 0; r < 8; ++r) {
      float v = fmaxf(s[0][r], s[1][r]);
      v = fmaxf(v, __shfl_xor(v, 1, 32));
      v = fmaxf(v, __shfl_xor(v, 2, 32));
      v = fmaxf(v, __shfl_xor(v, 4, 32));
      v = fmaxf(v, __shfl_xor(v, 8, 32));
      const float nm = fmaxf(Mr[r], v);
      alpha[r] = __expf(Mr[r] - nm);
      Mr[r] = nm;
      s[0][r] = __expf(s[0][r] - nm);
      s[1][r] = __expf(s[1][r] - nm);
      float rs = s[0][r] + s[1][r];
      rs += __shfl_xor(rs, 1, 32);
      rs += __shfl_xor(rs, 2, 32);
      rs += __shfl_xor(rs, 4, 32);
      rs += __shfl_xor(rs, 8, 32);
      Lr[r] = Lr[r] * alpha[r] + rs;
    }
    #pragma unroll
    for (int d = 0; d < 4; ++d)
      #pragma unroll
      for (int r = 0; r < 8; ++r)
        Oacc[d][r] *= alpha[r];

    // ---- P: C-layout -> f16 LDS -> A-fragment (wave-private scratch) ----
    #pragma unroll
    for (int nt = 0; nt < 2; ++nt)
      #pragma unroll
      for (int r = 0; r < 8; ++r) {
        const int m = r + hi16 * 8;
        const int n = nt * 16 + ln;
        myP[m * 32 + n] = (_Float16)s[nt][r];
      }
    asm volatile("s_wait_dscnt 0" ::: "memory");   // cross-lane LDS RAW in wave
    v16h pfrag;
    {
      const _Float16* rp = myP + ln * 32;
      const int base = hi16 ? 8 : 0;
      pfrag = cat8(*(const h8*)(rp + base), *(const h8*)(rp + base + 16));
    }

    // ---- O += P . V ----
    #pragma unroll
    for (int dt = 0; dt < 4; ++dt) {
      const _Float16* vp = Vb + (size_t)(dt * 16 + ln) * T + kv0 + (hi16 ? 8 : 0);
      v16h vfrag = cat8(*(const h8*)vp, *(const h8*)(vp + 16));
      Oacc[dt] = WMMA_F32_F16(pfrag, vfrag, Oacc[dt]);
    }
  }

  // ---- normalize and write attention output as f16 [b, t, h*Dh + d] ----
  const int b = bh >> 4, hh = bh & 15;
  #pragma unroll
  for (int r = 0; r < 8; ++r) {
    const float linv = 1.f / Lr[r];
    const int qi = q0 + r + hi16 * 8;
    _Float16* op = O16 + ((size_t)(b * T + qi)) * D + hh * Dh;
    #pragma unroll
    for (int dt = 0; dt < 4; ++dt)
      op[dt * 16 + ln] = (_Float16)(Oacc[dt][r] * linv);
  }
}

// ---------------------------------------------------------------------------
// Kernel 3: out = O16 @ Wo^T, fp32 output.  Same double-buffered NT GEMM.
// ---------------------------------------------------------------------------
__global__ __launch_bounds__(256)
void out_proj_kernel(const _Float16* __restrict__ A,   // (B*T, D) f16
                     const float* __restrict__ Wo,
                     float* __restrict__ out) {
  __shared__ __align__(16) _Float16 As[2][128 * 64];
  __shared__ __align__(16) _Float16 Ws[2][128 * 64];

  const int n0 = blockIdx.x * 128;
  const int m0 = blockIdx.y * 128;

  const int tid  = threadIdx.x;
  const int lane = tid & 31;
  const int wid  = tid >> 5;
  const int rm   = (wid >> 1) * 32;
  const int cn   = (wid & 1) * 64;
  const int hi16 = (lane >= 16) ? 1 : 0;
  const int ln   = lane & 15;

  v8f acc[2][4];
  #pragma unroll
  for (int i = 0; i < 2; ++i)
    #pragma unroll
    for (int j = 0; j < 4; ++j)
      acc[i][j] = v8f{0.f,0.f,0.f,0.f,0.f,0.f,0.f,0.f};

  h8     ra[4];   // A tile is already f16: 128x64 = 1024 h8 units, 4/thread
  float4 rw[8];

  #define FETCH_TILES2(K0)                                                    \
    {                                                                         \
      _Pragma("unroll")                                                       \
      for (int it = 0; it < 4; ++it) {                                        \
        const int idx = tid + it * 256;            /* 1024 h8 units */        \
        const int row = idx >> 3;                                             \
        const int c8  = (idx & 7) * 8;                                        \
        ra[it] = *(const h8*)(A + (size_t)(m0 + row) * D + (K0) + c8);        \
      }                                                                       \
      _Pragma("unroll")                                                       \
      for (int it = 0; it < 8; ++it) {                                        \
        const int idx = tid + it * 256;            /* 2048 float4 units */    \
        const int row = idx >> 4;                                             \
        const int c4  = (idx & 15) * 4;                                       \
        rw[it] = *(const float4*)(Wo + (size_t)(n0 + row) * D + (K0) + c4);   \
      }                                                                       \
    }

  #define STORE_TILES2(P)                                                     \
    {                                                                         \
      _Pragma("unroll")                                                       \
      for (int it = 0; it < 4; ++it) {                                        \
        const int idx = tid + it * 256;                                       \
        const int row = idx >> 3;                                             \
        const int c8  = (idx & 7) * 8;                                        \
        *(h8*)&As[P][row * 64 + c8] = ra[it];                                 \
      }                                                                       \
      _Pragma("unroll")                                                       \
      for (int it = 0; it < 8; ++it) {                                        \
        const int idx = tid + it * 256;                                       \
        const int row = idx >> 4;                                             \
        const int c4  = (idx & 15) * 4;                                       \
        _Float16* dw = &Ws[P][row * 64 + c4];                                 \
        dw[0] = (_Float16)rw[it].x; dw[1] = (_Float16)rw[it].y;               \
        dw[2] = (_Float16)rw[it].z; dw[3] = (_Float16)rw[it].w;               \
      }                                                                       \
    }

  #define COMPUTE_STAGE2(P)                                                   \
    {                                                                         \
      v16h afrag[2][2];                                                       \
      _Pragma("unroll")                                                       \
      for (int mt = 0; mt < 2; ++mt) {                                        \
        const _Float16* rp = &As[P][(rm + mt * 16 + ln) * 64];                \
        _Pragma("unroll")                                                     \
        for (int kc = 0; kc < 2; ++kc) {                                      \
          const int base = kc * 32 + (hi16 ? 8 : 0);                          \
          afrag[mt][kc] = cat8(*(const h8*)(rp + base),                       \
                               *(const h8*)(rp + base + 16));                 \
        }                                                                     \
      }                                                                       \
      _Pragma("unroll")                                                       \
      for (int kc = 0; kc < 2; ++kc)                                          \
        _Pragma("unroll")                                                     \
        for (int nt = 0; nt < 4; ++nt) {                                      \
          const _Float16* rp = &Ws[P][(cn + nt * 16 + ln) * 64];              \
          const int base = kc * 32 + (hi16 ? 8 : 0);                          \
          v16h bfrag = cat8(*(const h8*)(rp + base),                          \
                            *(const h8*)(rp + base + 16));                    \
          acc[0][nt] = WMMA_F32_F16(afrag[0][kc], bfrag, acc[0][nt]);         \
          acc[1][nt] = WMMA_F32_F16(afrag[1][kc], bfrag, acc[1][nt]);         \
        }                                                                     \
    }

  FETCH_TILES2(0);
  STORE_TILES2(0);
  __syncthreads();

  int p = 0;
  for (int k0 = 0;;) {
    const bool more = (k0 + 64) < D;
    if (more) FETCH_TILES2(k0 + 64);
    COMPUTE_STAGE2(p);
    if (!more) break;
    STORE_TILES2(p ^ 1);
    __syncthreads();
    p ^= 1;
    k0 += 64;
  }

  #pragma unroll
  for (int mt = 0; mt < 2; ++mt)
    #pragma unroll
    for (int nt = 0; nt < 4; ++nt)
      #pragma unroll
      for (int r = 0; r < 8; ++r) {
        const int m = m0 + rm + mt * 16 + r + hi16 * 8;
        const int n = n0 + cn + nt * 16 + ln;
        out[(size_t)m * D + n] = acc[mt][nt][r];
      }
  #undef FETCH_TILES2
  #undef STORE_TILES2
  #undef COMPUTE_STAGE2
}

// ---------------------------------------------------------------------------
extern "C" void kernel_launch(void* const* d_in, const int* in_sizes, int n_in,
                              void* d_out, int out_size, void* d_ws, size_t ws_size,
                              hipStream_t stream) {
  (void)in_sizes; (void)n_in; (void)out_size; (void)ws_size;
  const float* x  = (const float*)d_in[0];
  const float* Wq = (const float*)d_in[1];
  const float* Wk = (const float*)d_in[2];
  const float* Wv = (const float*)d_in[3];
  const float* Wo = (const float*)d_in[4];
  float* out = (float*)d_out;

  char* ws = (char*)d_ws;
  const size_t SEG = (size_t)8 * 1024 * 1024;   // 8 MB each
  _Float16* Qp  = (_Float16*)(ws + 0 * SEG);    // [b,h,t,dh] f16, pre-scaled
  _Float16* Kp  = (_Float16*)(ws + 1 * SEG);    // [b,h,t,dh] f16
  _Float16* Vtp = (_Float16*)(ws + 2 * SEG);    // [b,h,dh,t] f16 (transposed)
  _Float16* Op  = (_Float16*)(ws + 3 * SEG);    // [b,t,h*dh] f16

  proj_qkv_kernel<<<dim3(8, 32, 3), 256, 0, stream>>>(x, Wq, Wk, Wv, Qp, Kp, Vtp);
  flash_attn_kernel<<<dim3(1024), 128, 0, stream>>>(Qp, Kp, Vtp, Op);
  out_proj_kernel<<<dim3(8, 32), 256, 0, stream>>>(Op, Wo, out);
}
